// FusedLlamaLowBitDecoderlayer_57535381897585
// MI455X (gfx1250) — compile-verified, matrix operand order checked by reading, not run on hardware
//
#include <hip/hip_runtime.h>
#include <hip/hip_bf16.h>
#include <math.h>

// ---------------------------------------------------------------------------
// Types for CDNA5 WMMA (wave32, v_wmma_f32_16x16x32_f16)
// ---------------------------------------------------------------------------
typedef _Float16 half_t;
typedef __attribute__((ext_vector_type(16))) _Float16 v16h;
typedef __attribute__((ext_vector_type(8)))  _Float16 v8h;
typedef __attribute__((ext_vector_type(8)))  float    v8f;

union HFrag { v16h v; v8h h[2]; };

#define B_SZ 64
#define T_SZ 218
#define D_SZ 512
#define M_SZ (B_SZ * T_SZ)   // 13952 = 128 * 109
#define TPAD 224             // keys padded to multiple of 32
#define NHEAD 4
#define DK 128
#define FFN_D 2048
#define KTAP 11
#define PADC 5
#define TROWS (T_SZ + 2 * PADC)      // 228 padded rows per batch
#define KFSMN (KTAP * D_SZ)          // 5632 contiguous K for fsmn GEMM

__device__ __forceinline__ v8f wmma32(v16h a, v16h b, v8f c) {
  return __builtin_amdgcn_wmma_f32_16x16x32_f16(false, a, false, b, (short)0, c,
                                                false, false);
}

// A/B fragment for 16x16x32 f16 WMMA.
// lane L holds row/col (L&15); kbase = (L<16)?0:8; halves:
//   elements 0..7  = K kbase..kbase+7
//   elements 8..15 = K kbase+16..kbase+23
__device__ __forceinline__ v16h load_frag(const half_t* p, int kb) {
  HFrag f;
  f.h[0] = *(const v8h*)(p + kb);
  f.h[1] = *(const v8h*)(p + kb + 16);
  return f.v;
}

// Wave-level GEMM: accumulate a 32(M) x 64(N) tile (8 WMMAs / 12 16B-loads per
// k-step). Branchless: A rows must be contiguous over K.
__device__ __forceinline__ void gemm_32x64(const half_t* a0, const half_t* a1,
                                           const half_t* bt, int K, int n0,
                                           int lane, v8f acc[8]) {
  const int kb = (lane < 16) ? 0 : 8;
  const int bcol = n0 + (lane & 15);
  const half_t* b0 = bt + (size_t)(bcol +  0) * K;
  const half_t* b1 = bt + (size_t)(bcol + 16) * K;
  const half_t* b2 = bt + (size_t)(bcol + 32) * K;
  const half_t* b3 = bt + (size_t)(bcol + 48) * K;
  for (int k0 = 0; k0 < K; k0 += 32) {
    v16h aA = load_frag(a0 + k0, kb);
    v16h aB = load_frag(a1 + k0, kb);
    v16h bf0 = load_frag(b0 + k0, kb);
    v16h bf1 = load_frag(b1 + k0, kb);
    v16h bf2 = load_frag(b2 + k0, kb);
    v16h bf3 = load_frag(b3 + k0, kb);
    acc[0] = wmma32(aA, bf0, acc[0]);
    acc[1] = wmma32(aA, bf1, acc[1]);
    acc[2] = wmma32(aA, bf2, acc[2]);
    acc[3] = wmma32(aA, bf3, acc[3]);
    acc[4] = wmma32(aB, bf0, acc[4]);
    acc[5] = wmma32(aB, bf1, acc[5]);
    acc[6] = wmma32(aB, bf2, acc[6]);
    acc[7] = wmma32(aB, bf3, acc[7]);
  }
}

// C fragment mapping (32-bit 16x16): element v of lane L is (m = v + 8*(L>>4), n = L&15)
#define C_ROW(vv, lane) ((vv) + (((lane) >> 4) << 3))
#define C_COL(lane) ((lane) & 15)

// 8-wave block: 4(M) x 2(N) waves -> 128 x 128 block tile
#define GEMM_PROLOGUE()                                       \
  const int lane = threadIdx.x & 31;                          \
  const int wave = threadIdx.x >> 5;                          \
  const int row0 = blockIdx.x * 128 + (wave & 3) * 32;        \
  const int n0 = blockIdx.y * 128 + (wave >> 2) * 64;         \
  v8f acc[8] = {{}, {}, {}, {}, {}, {}, {}, {}}

// ---------------------------------------------------------------------------
// Weight conversion / repack kernels (fp32 -> f16, transposed to [N][K])
// ---------------------------------------------------------------------------
__global__ void k_transpose_w(const float* __restrict__ w, half_t* __restrict__ wt,
                              int K, int N) {
  size_t i = (size_t)blockIdx.x * 256 + threadIdx.x;
  if (i >= (size_t)K * N) return;
  int n = (int)(i / K);
  int k = (int)(i % K);
  wt[i] = (half_t)w[(size_t)k * N + n];
}

// fsmn_w (out=512, in=512, tap=11) -> wt[out][tap][in] f16  (row out contiguous
// over K = 11*512 so the conv becomes one branchless GEMM)
__global__ void k_repack_fsmn(const float* __restrict__ w, half_t* __restrict__ wt) {
  size_t i = (size_t)blockIdx.x * 256 + threadIdx.x;
  if (i >= (size_t)D_SZ * KFSMN) return;
  int o = (int)(i / KFSMN);
  int rem = (int)(i % KFSMN);
  int tap = rem / D_SZ;
  int in = rem % D_SZ;
  wt[i] = (half_t)w[((size_t)o * D_SZ + in) * KTAP + tap];
}

// vmpad[b][j][c] = (5 <= j < 223) ? vm[b][j-5][c] : 0   (TROWS=228 rows/batch)
__global__ void k_build_vmpad(const half_t* __restrict__ vm, half_t* __restrict__ vmpad) {
  size_t i = (size_t)blockIdx.x * 256 + threadIdx.x;
  if (i >= (size_t)B_SZ * TROWS * D_SZ) return;
  int c = (int)(i % D_SZ);
  size_t r = i / D_SZ;
  int j = (int)(r % TROWS);
  int b = (int)(r / TROWS);
  int st = j - PADC;
  half_t val = (half_t)0;
  if (st >= 0 && st < T_SZ) val = vm[(size_t)(b * T_SZ + st) * D_SZ + c];
  vmpad[i] = val;
}

// ---------------------------------------------------------------------------
// LayerNorm: one wave per row of 512, writes f16
// ---------------------------------------------------------------------------
__global__ void k_layernorm(const float* __restrict__ x, const float* __restrict__ w,
                            const float* __restrict__ b, half_t* __restrict__ out) {
  int row = blockIdx.x * 8 + (threadIdx.x >> 5);
  int lane = threadIdx.x & 31;
  const float* xr = x + (size_t)row * D_SZ;
  float s = 0.f, ss = 0.f;
#pragma unroll
  for (int j = 0; j < 16; ++j) {
    float v = xr[lane + 32 * j];
    s += v; ss += v * v;
  }
#pragma unroll
  for (int off = 16; off >= 1; off >>= 1) {
    s += __shfl_xor(s, off, 32);
    ss += __shfl_xor(ss, off, 32);
  }
  float mean = s * (1.f / D_SZ);
  float var = ss * (1.f / D_SZ) - mean * mean;
  float inv = rsqrtf(var + 1e-5f);
  half_t* orow = out + (size_t)row * D_SZ;
#pragma unroll
  for (int j = 0; j < 16; ++j) {
    int i = lane + 32 * j;
    orow[i] = (half_t)((xr[i] - mean) * inv * w[i] + b[i]);
  }
}

// ---------------------------------------------------------------------------
// QKV GEMM: h0[M,512] @ qkv_wT[1536,512] + bias -> q,k,v,vm (all f16 [M,512])
// ---------------------------------------------------------------------------
__global__ void k_gemm_qkv(const half_t* __restrict__ A, const half_t* __restrict__ BT,
                           const float* __restrict__ bias, const float* __restrict__ masks,
                           half_t* __restrict__ q, half_t* __restrict__ k,
                           half_t* __restrict__ v, half_t* __restrict__ vm) {
  GEMM_PROLOGUE();
  const half_t* a0 = A + (size_t)(row0 + (lane & 15)) * D_SZ;
  const half_t* a1 = a0 + (size_t)16 * D_SZ;
  gemm_32x64(a0, a1, BT, D_SZ, n0, lane, acc);
#pragma unroll
  for (int h2 = 0; h2 < 2; ++h2) {
#pragma unroll
    for (int s = 0; s < 4; ++s) {
#pragma unroll
      for (int vv = 0; vv < 8; ++vv) {
        int rg = row0 + h2 * 16 + C_ROW(vv, lane);
        int n = n0 + s * 16 + C_COL(lane);
        float val = acc[h2 * 4 + s][vv] + bias[n];
        size_t idx = (size_t)rg * D_SZ + (n & 511);
        if (n < 512) {
          q[idx] = (half_t)val;
        } else if (n < 1024) {
          k[idx] = (half_t)val;
        } else {
          v[idx] = (half_t)val;
          vm[idx] = (half_t)(val * masks[rg]);
        }
      }
    }
  }
}

// ---------------------------------------------------------------------------
// FSMN: fsmn = (Conv1d(vm, w, same) + vm) * mask, as ONE GEMM with K=5632:
// A row for (b,t) = vmpad[b][t..t+10][:] (contiguous), B = wt[out][tap][in]
// ---------------------------------------------------------------------------
__global__ void k_gemm_fsmn(const half_t* __restrict__ vmpad, const half_t* __restrict__ WT,
                            const half_t* __restrict__ vm, const float* __restrict__ masks,
                            float* __restrict__ fsmn) {
  GEMM_PROLOGUE();
  int rA = row0 + (lane & 15);
  int rB = rA + 16;
  int bA = rA / T_SZ, tA = rA - bA * T_SZ;
  int bB = rB / T_SZ, tB = rB - bB * T_SZ;
  const half_t* a0 = vmpad + (size_t)(bA * TROWS + tA) * D_SZ;
  const half_t* a1 = vmpad + (size_t)(bB * TROWS + tB) * D_SZ;
  gemm_32x64(a0, a1, WT, KFSMN, n0, lane, acc);
#pragma unroll
  for (int h2 = 0; h2 < 2; ++h2) {
#pragma unroll
    for (int s = 0; s < 4; ++s) {
#pragma unroll
      for (int vv = 0; vv < 8; ++vv) {
        int rg = row0 + h2 * 16 + C_ROW(vv, lane);
        int n = n0 + s * 16 + C_COL(lane);
        size_t idx = (size_t)rg * D_SZ + n;
        fsmn[idx] = (acc[h2 * 4 + s][vv] + (float)vm[idx]) * masks[rg];
      }
    }
  }
}

// ---------------------------------------------------------------------------
// Build vT[b][h][d][key] f16 with zero padding to TPAD keys
// ---------------------------------------------------------------------------
__global__ void k_build_vT(const half_t* __restrict__ v, half_t* __restrict__ vT) {
  size_t i = (size_t)blockIdx.x * 256 + threadIdx.x;
  if (i >= (size_t)B_SZ * NHEAD * DK * TPAD) return;
  int key = (int)(i % TPAD);
  size_t r = i / TPAD;
  int d = (int)(r % DK);
  size_t r2 = r / DK;
  int h = (int)(r2 % NHEAD);
  int b = (int)(r2 / NHEAD);
  half_t val = (half_t)0;
  if (key < T_SZ) val = v[(size_t)(b * T_SZ + key) * D_SZ + h * DK + d];
  vT[i] = val;
}

// ---------------------------------------------------------------------------
// Attention: one wave per (b, h, 16-query tile). scores->softmax->ctx via WMMA
// ---------------------------------------------------------------------------
__global__ void k_attention(const half_t* __restrict__ q, const half_t* __restrict__ k,
                            const half_t* __restrict__ vT, const float* __restrict__ masks,
                            half_t* __restrict__ ctx) {
  __shared__ __align__(16) float s_sc[16 * TPAD];
  __shared__ __align__(16) half_t s_at[16 * TPAD];
  const int lane = threadIdx.x & 31;
  const int kb = (lane < 16) ? 0 : 8;
  int blk = blockIdx.x;
  int qt = blk % 14;
  int bh = blk / 14;
  int h = bh & (NHEAD - 1);
  int b = bh >> 2;

  // ---- scores = q @ k^T / sqrt(dk) ----
  int qrl = qt * 16 + (lane & 15);
  int qrc = (qrl < T_SZ) ? qrl : (T_SZ - 1);
  const half_t* arow = q + (size_t)(b * T_SZ + qrc) * D_SZ + h * DK;
  const float scale = 0.08838834764831845f;  // 1/sqrt(128)
  for (int kt = 0; kt < 14; ++kt) {
    v8f acc = {};
    int key = kt * 16 + (lane & 15);
    int keyc = (key < T_SZ) ? key : (T_SZ - 1);  // clamp; masked in softmax
    const half_t* brow = k + (size_t)(b * T_SZ + keyc) * D_SZ + h * DK;
#pragma unroll
    for (int k0 = 0; k0 < DK; k0 += 32) {
      acc = wmma32(load_frag(arow + k0, kb), load_frag(brow + k0, kb), acc);
    }
#pragma unroll
    for (int vv = 0; vv < 8; ++vv) {
      int m = C_ROW(vv, lane);
      s_sc[m * TPAD + kt * 16 + C_COL(lane)] = acc[vv] * scale;
    }
  }
  __syncthreads();

  // ---- masked softmax (lane handles row = lane&15, column half = lane>>4) ----
  int row = lane & 15;
  int c0 = (lane >> 4) * (TPAD / 2);
  const float NEG = -3.0e38f;
  float mx = NEG;
  for (int j = 0; j < TPAD / 2; ++j) {
    int key = c0 + j;
    float val = (key < T_SZ && masks[b * T_SZ + key] > 0.f) ? s_sc[row * TPAD + key] : NEG;
    s_sc[row * TPAD + key] = val;
    mx = fmaxf(mx, val);
  }
  mx = fmaxf(mx, __shfl_xor(mx, 16, 32));
  float sum = 0.f;
  for (int j = 0; j < TPAD / 2; ++j) {
    int key = c0 + j;
    float val = s_sc[row * TPAD + key];
    float e = (val <= 0.5f * NEG) ? 0.f : __expf(val - mx);
    s_sc[row * TPAD + key] = e;
    sum += e;
  }
  sum += __shfl_xor(sum, 16, 32);
  float rinv = 1.f / sum;
  for (int j = 0; j < TPAD / 2; ++j) {
    int key = c0 + j;
    s_at[row * TPAD + key] = (half_t)(s_sc[row * TPAD + key] * rinv);
  }
  __syncthreads();

  // ---- ctx = attn @ v  (B fragment from vT[b][h][d][key]) ----
  const half_t* arow2 = s_at + (size_t)(lane & 15) * TPAD;
  for (int nt = 0; nt < DK / 16; ++nt) {
    v8f acc = {};
    int n = nt * 16 + (lane & 15);
    const half_t* brow = vT + (size_t)((b * NHEAD + h) * DK + n) * TPAD;
#pragma unroll
    for (int k0 = 0; k0 < TPAD; k0 += 32) {
      acc = wmma32(load_frag(arow2 + k0, kb), load_frag(brow + k0, kb), acc);
    }
#pragma unroll
    for (int vv = 0; vv < 8; ++vv) {
      int m = C_ROW(vv, lane);
      int tq = qt * 16 + m;
      if (tq < T_SZ)
        ctx[(size_t)(b * T_SZ + tq) * D_SZ + h * DK + nt * 16 + C_COL(lane)] =
            (half_t)acc[vv];
    }
  }
}

// ---------------------------------------------------------------------------
// Out projection: x1 = x + ctx @ out_wT + out_b + fsmn
// ---------------------------------------------------------------------------
__global__ void k_gemm_out(const half_t* __restrict__ A, const half_t* __restrict__ BT,
                           const float* __restrict__ bias, const float* __restrict__ fsmn,
                           const float* __restrict__ x, float* __restrict__ x1) {
  GEMM_PROLOGUE();
  const half_t* a0 = A + (size_t)(row0 + (lane & 15)) * D_SZ;
  const half_t* a1 = a0 + (size_t)16 * D_SZ;
  gemm_32x64(a0, a1, BT, D_SZ, n0, lane, acc);
#pragma unroll
  for (int h2 = 0; h2 < 2; ++h2) {
#pragma unroll
    for (int s = 0; s < 4; ++s) {
#pragma unroll
      for (int vv = 0; vv < 8; ++vv) {
        int rg = row0 + h2 * 16 + C_ROW(vv, lane);
        int n = n0 + s * 16 + C_COL(lane);
        size_t idx = (size_t)rg * D_SZ + n;
        x1[idx] = acc[h2 * 4 + s][vv] + bias[n] + fsmn[idx] + x[idx];
      }
    }
  }
}

// ---------------------------------------------------------------------------
// FFN1: mid = relu(h1 @ w1T + b1)  (f16 out, N=2048)
// ---------------------------------------------------------------------------
__global__ void k_gemm_ffn1(const half_t* __restrict__ A, const half_t* __restrict__ BT,
                            const float* __restrict__ bias, half_t* __restrict__ mid) {
  GEMM_PROLOGUE();
  const half_t* a0 = A + (size_t)(row0 + (lane & 15)) * D_SZ;
  const half_t* a1 = a0 + (size_t)16 * D_SZ;
  gemm_32x64(a0, a1, BT, D_SZ, n0, lane, acc);
#pragma unroll
  for (int h2 = 0; h2 < 2; ++h2) {
#pragma unroll
    for (int s = 0; s < 4; ++s) {
#pragma unroll
      for (int vv = 0; vv < 8; ++vv) {
        int rg = row0 + h2 * 16 + C_ROW(vv, lane);
        int n = n0 + s * 16 + C_COL(lane);
        float val = acc[h2 * 4 + s][vv] + bias[n];
        mid[(size_t)rg * FFN_D + n] = (half_t)(val > 0.f ? val : 0.f);
      }
    }
  }
}

// ---------------------------------------------------------------------------
// FFN2: out = x1 + mid @ w2T + b2  (K=2048, fp32 out)
// ---------------------------------------------------------------------------
__global__ void k_gemm_ffn2(const half_t* __restrict__ A, const half_t* __restrict__ BT,
                            const float* __restrict__ bias, const float* __restrict__ x1,
                            float* __restrict__ out) {
  GEMM_PROLOGUE();
  const half_t* a0 = A + (size_t)(row0 + (lane & 15)) * FFN_D;
  const half_t* a1 = a0 + (size_t)16 * FFN_D;
  gemm_32x64(a0, a1, BT, FFN_D, n0, lane, acc);
#pragma unroll
  for (int h2 = 0; h2 < 2; ++h2) {
#pragma unroll
    for (int s = 0; s < 4; ++s) {
#pragma unroll
      for (int vv = 0; vv < 8; ++vv) {
        int rg = row0 + h2 * 16 + C_ROW(vv, lane);
        int n = n0 + s * 16 + C_COL(lane);
        size_t idx = (size_t)rg * D_SZ + n;
        out[idx] = acc[h2 * 4 + s][vv] + bias[n] + x1[idx];
      }
    }
  }
}

// ---------------------------------------------------------------------------
// Host-side launch
// ---------------------------------------------------------------------------
static inline size_t align_up256(size_t x) { return (x + 255) & ~(size_t)255; }

extern "C" void kernel_launch(void* const* d_in, const int* in_sizes, int n_in,
                              void* d_out, int out_size, void* d_ws, size_t ws_size,
                              hipStream_t stream) {
  const float* x      = (const float*)d_in[0];
  const float* masks  = (const float*)d_in[1];
  const float* ln0_w  = (const float*)d_in[2];
  const float* ln0_b  = (const float*)d_in[3];
  const float* ln1_w  = (const float*)d_in[4];
  const float* ln1_b  = (const float*)d_in[5];
  const float* qkv_w  = (const float*)d_in[6];
  const float* qkv_b  = (const float*)d_in[7];
  const float* out_w  = (const float*)d_in[8];
  const float* out_b  = (const float*)d_in[9];
  const float* fsmn_w = (const float*)d_in[10];
  const float* w1     = (const float*)d_in[11];
  const float* b1     = (const float*)d_in[12];
  const float* w2     = (const float*)d_in[13];
  const float* b2     = (const float*)d_in[14];
  float* out = (float*)d_out;

  // workspace carve-out
  char* ws = (char*)d_ws;
  size_t cur = 0;
  auto alloc = [&](size_t bytes) {
    void* p = ws + cur;
    cur += align_up256(bytes);
    return p;
  };
  const size_t MD2 = (size_t)M_SZ * D_SZ * sizeof(half_t);
  half_t* h0      = (half_t*)alloc(MD2);                       // LN output (reused for LN1)
  half_t* qbuf    = (half_t*)alloc(MD2);
  half_t* kbuf    = (half_t*)alloc(MD2);
  half_t* vbuf    = (half_t*)alloc(MD2);
  half_t* vmbuf   = (half_t*)alloc(MD2);
  half_t* vmpad   = (half_t*)alloc((size_t)B_SZ * TROWS * D_SZ * sizeof(half_t));
  half_t* vT      = (half_t*)alloc((size_t)B_SZ * NHEAD * DK * TPAD * sizeof(half_t));
  half_t* ctx     = (half_t*)alloc(MD2);
  float*  fsmn    = (float*)alloc((size_t)M_SZ * D_SZ * sizeof(float));
  float*  x1      = (float*)alloc((size_t)M_SZ * D_SZ * sizeof(float));
  half_t* mid     = (half_t*)alloc((size_t)M_SZ * FFN_D * sizeof(half_t));
  half_t* qkv_wT  = (half_t*)alloc((size_t)1536 * D_SZ * sizeof(half_t));
  half_t* out_wT  = (half_t*)alloc((size_t)D_SZ * D_SZ * sizeof(half_t));
  half_t* fsmn_wT = (half_t*)alloc((size_t)D_SZ * KFSMN * sizeof(half_t));
  half_t* w1T     = (half_t*)alloc((size_t)FFN_D * D_SZ * sizeof(half_t));
  half_t* w2T     = (half_t*)alloc((size_t)D_SZ * FFN_D * sizeof(half_t));
  (void)ws_size; (void)n_in; (void)in_sizes; (void)out_size;

  // 1) weight conversion (fp32 -> f16, transposed to [N][K])
  {
    size_t n;
    n = (size_t)D_SZ * 1536;
    k_transpose_w<<<dim3((unsigned)((n + 255) / 256)), 256, 0, stream>>>(qkv_w, qkv_wT, D_SZ, 1536);
    n = (size_t)D_SZ * D_SZ;
    k_transpose_w<<<dim3((unsigned)((n + 255) / 256)), 256, 0, stream>>>(out_w, out_wT, D_SZ, D_SZ);
    n = (size_t)D_SZ * FFN_D;
    k_transpose_w<<<dim3((unsigned)((n + 255) / 256)), 256, 0, stream>>>(w1, w1T, D_SZ, FFN_D);
    n = (size_t)FFN_D * D_SZ;
    k_transpose_w<<<dim3((unsigned)((n + 255) / 256)), 256, 0, stream>>>(w2, w2T, FFN_D, D_SZ);
    n = (size_t)D_SZ * KFSMN;
    k_repack_fsmn<<<dim3((unsigned)((n + 255) / 256)), 256, 0, stream>>>(fsmn_w, fsmn_wT);
  }

  // 2) LN0
  k_layernorm<<<dim3(M_SZ / 8), 256, 0, stream>>>(x, ln0_w, ln0_b, h0);

  // 3) QKV GEMM (M x 1536, K=512), 128x128 block tiles
  k_gemm_qkv<<<dim3(M_SZ / 128, 1536 / 128), 256, 0, stream>>>(h0, qkv_wT, qkv_b, masks,
                                                               qbuf, kbuf, vbuf, vmbuf);

  // 4a) padded vm for branchless fsmn GEMM
  {
    size_t n = (size_t)B_SZ * TROWS * D_SZ;
    k_build_vmpad<<<dim3((unsigned)((n + 255) / 256)), 256, 0, stream>>>(vmbuf, vmpad);
  }
  // 4b) vT for attention's second GEMM
  {
    size_t n = (size_t)B_SZ * NHEAD * DK * TPAD;
    k_build_vT<<<dim3((unsigned)((n + 255) / 256)), 256, 0, stream>>>(vbuf, vT);
  }

  // 5) FSMN conv as one K=5632 GEMM
  k_gemm_fsmn<<<dim3(M_SZ / 128, D_SZ / 128), 256, 0, stream>>>(vmpad, fsmn_wT, vmbuf,
                                                                masks, fsmn);

  // 6) attention (one wave per (b,h,qtile))
  k_attention<<<dim3(B_SZ * NHEAD * 14), 32, 0, stream>>>(qbuf, kbuf, vT, masks, ctx);

  // 7) out projection + residual + fsmn
  k_gemm_out<<<dim3(M_SZ / 128, D_SZ / 128), 256, 0, stream>>>(ctx, out_wT, out_b, fsmn, x, x1);

  // 8) LN1 (reuse h0)
  k_layernorm<<<dim3(M_SZ / 8), 256, 0, stream>>>(x1, ln1_w, ln1_b, h0);

  // 9) FFN1 (relu)
  k_gemm_ffn1<<<dim3(M_SZ / 128, FFN_D / 128), 256, 0, stream>>>(h0, w1T, b1, mid);

  // 10) FFN2 + residual -> d_out
  k_gemm_ffn2<<<dim3(M_SZ / 128, D_SZ / 128), 256, 0, stream>>>(mid, w2T, b2, x1, out);
}
